// GraphSAGE_74345883894268
// MI455X (gfx1250) — matrix operand order repaired and checked by reading, not compiled, not run on hardware
//
#include <hip/hip_runtime.h>
#include <hip/hip_bf16.h>

// ---------------------------------------------------------------------------
// GraphSAGE (2-layer, mean aggregation) for MI455X / gfx1250
//   layer: out = (agg/deg) @ Wl.T + b + x @ Wr.T   (ReLU after layer 1)
// GEMMs on v_wmma_f32_16x16x32_bf16 with ALL f32->bf16 conversion hoisted out
// of the WMMA loop (weights/x/mean pre-converted; h1 produced in bf16).
// Scatter via f32 atomics; all buffers L2-resident (<192MB).
// ---------------------------------------------------------------------------

#define GS_IN   128
#define GS_HID  256
#define GS_OUT  128

typedef __attribute__((ext_vector_type(16))) __bf16 v16bf;
typedef __attribute__((ext_vector_type(8)))  __bf16 v8bf;
typedef __attribute__((ext_vector_type(2)))  __bf16 v2bf;
typedef __attribute__((ext_vector_type(8)))  float  v8f;

// round-to-nearest-even f32 -> bf16, scalar (epilogue only)
__device__ __forceinline__ __bf16 f32_to_bf16_rne(float f) {
    unsigned u = __builtin_bit_cast(unsigned, f);
    unsigned r = (u + 0x7FFFu + ((u >> 16) & 1u)) >> 16;
    unsigned short s = (unsigned short)r;
    return __builtin_bit_cast(__bf16, s);
}

// RNE pack of two f32 into one dword of two bf16 via v_perm_b32
__device__ __forceinline__ v2bf pack_bf16(float lo, float hi) {
    unsigned ulo = __builtin_bit_cast(unsigned, lo);
    unsigned uhi = __builtin_bit_cast(unsigned, hi);
    ulo += 0x7FFFu + ((ulo >> 16) & 1u);
    uhi += 0x7FFFu + ((uhi >> 16) & 1u);
    // dest = { uhi[31:16], ulo[31:16] }  (byte sel: src1 bytes 2,3 ; src0 bytes 2,3)
    unsigned packed = __builtin_amdgcn_perm(uhi, ulo, 0x07060302u);
    return __builtin_bit_cast(v2bf, packed);
}

__device__ __forceinline__ float bf16_bits_lo_to_f32(unsigned u) {
    return __builtin_bit_cast(float, u << 16);
}
__device__ __forceinline__ float bf16_bits_hi_to_f32(unsigned u) {
    return __builtin_bit_cast(float, u & 0xFFFF0000u);
}

// ---------------------------------------------------------------------------
// Zero a float buffer (float4 grid-stride).
// ---------------------------------------------------------------------------
__global__ void gs_zero_kernel(float4* __restrict__ p, long n4) {
    long i = (long)blockIdx.x * blockDim.x + threadIdx.x;
    long stride = (long)gridDim.x * blockDim.x;
    float4 z = make_float4(0.f, 0.f, 0.f, 0.f);
    for (; i < n4; i += stride) p[i] = z;
}

// ---------------------------------------------------------------------------
// f32 -> bf16 convert (pairwise, grid-stride). n2 = element-count / 2.
// ---------------------------------------------------------------------------
__global__ void gs_cvt_bf16_kernel(const float* __restrict__ src,
                                   __bf16* __restrict__ dst, long n2) {
    long i = (long)blockIdx.x * blockDim.x + threadIdx.x;
    long stride = (long)gridDim.x * blockDim.x;
    const float2* s2 = (const float2*)src;
    v2bf* d2 = (v2bf*)dst;
    for (; i < n2; i += stride) {
        float2 v = s2[i];
        d2[i] = pack_bf16(v.x, v.y);
    }
}

// ---------------------------------------------------------------------------
// mean[m,k] = bf16( agg[m,k] / max(deg[m],1) )   (pairwise, grid-stride)
// ---------------------------------------------------------------------------
template <int D>
__global__ void gs_mean_kernel(const float* __restrict__ agg,
                               const float* __restrict__ deg,
                               __bf16* __restrict__ mbf, long n2) {
    long i = (long)blockIdx.x * blockDim.x + threadIdx.x;
    long stride = (long)gridDim.x * blockDim.x;
    const float2* a2 = (const float2*)agg;
    v2bf* d2 = (v2bf*)mbf;
    for (; i < n2; i += stride) {
        long m = (i * 2) / D;
        float invd = 1.0f / fmaxf(deg[m], 1.0f);
        float2 v = a2[i];
        d2[i] = pack_bf16(v.x * invd, v.y * invd);
    }
}

// ---------------------------------------------------------------------------
// Edge scatter, f32 source: agg[col[e],:] += x[row[e],:] (+deg if DO_DEG).
// D/4 threads per edge, float4 gather, 4 f32 atomics per thread.
// ---------------------------------------------------------------------------
template <int D, bool DO_DEG>
__global__ void gs_scatter_f32_kernel(const float* __restrict__ x,
                                      const int*   __restrict__ row,
                                      const int*   __restrict__ col,
                                      float* __restrict__ agg,
                                      float* __restrict__ deg,
                                      long nedges) {
    constexpr int LPE = D / 4;
    long tid  = (long)blockIdx.x * blockDim.x + threadIdx.x;
    long edge = tid / LPE;
    int  f4   = (int)(tid % LPE);
    if (edge >= nedges) return;

    int r = row[edge];
    int c = col[edge];

    float4 v = ((const float4*)(x + (long)r * D))[f4];
    float* dst = agg + (long)c * D + f4 * 4;
    atomicAdd(dst + 0, v.x);
    atomicAdd(dst + 1, v.y);
    atomicAdd(dst + 2, v.z);
    atomicAdd(dst + 3, v.w);

    if (DO_DEG && f4 == 0) atomicAdd(&deg[c], 1.0f);
}

// ---------------------------------------------------------------------------
// Edge scatter, bf16 source (halves gather traffic): D/8 threads per edge,
// one b128 gather of 8 bf16, shift-convert, 8 f32 atomics.
// ---------------------------------------------------------------------------
template <int D>
__global__ void gs_scatter_bf16_kernel(const __bf16* __restrict__ x,
                                       const int*    __restrict__ row,
                                       const int*    __restrict__ col,
                                       float* __restrict__ agg,
                                       long nedges) {
    constexpr int LPE = D / 8;
    long tid  = (long)blockIdx.x * blockDim.x + threadIdx.x;
    long edge = tid / LPE;
    int  f8   = (int)(tid % LPE);
    if (edge >= nedges) return;

    int r = row[edge];
    int c = col[edge];

    uint4 v = ((const uint4*)(x + (long)r * D))[f8];
    float* dst = agg + (long)c * D + f8 * 8;
    atomicAdd(dst + 0, bf16_bits_lo_to_f32(v.x));
    atomicAdd(dst + 1, bf16_bits_hi_to_f32(v.x));
    atomicAdd(dst + 2, bf16_bits_lo_to_f32(v.y));
    atomicAdd(dst + 3, bf16_bits_hi_to_f32(v.y));
    atomicAdd(dst + 4, bf16_bits_lo_to_f32(v.z));
    atomicAdd(dst + 5, bf16_bits_hi_to_f32(v.z));
    atomicAdd(dst + 6, bf16_bits_lo_to_f32(v.w));
    atomicAdd(dst + 7, bf16_bits_hi_to_f32(v.w));
}

// ---------------------------------------------------------------------------
// A-fragment loader (16x32 bf16, ISA 7.12.2):
// lane l owns row l&15; elem e -> k = ((e&8)<<1)|((l>>4)<<3)|(e&7)
// => two contiguous 8-elem runs at k0+laneHi*8 and k0+16+laneHi*8.
// ---------------------------------------------------------------------------
__device__ __forceinline__ v16bf load_a_frag(const __bf16* __restrict__ rowp,
                                             int k0, int laneHi) {
    v8bf lo = *(const v8bf*)(rowp + k0 + laneHi * 8);
    v8bf hi = *(const v8bf*)(rowp + k0 + 16 + laneHi * 8);
    return __builtin_shufflevector(lo, hi, 0, 1, 2, 3, 4, 5, 6, 7,
                                           8, 9, 10, 11, 12, 13, 14, 15);
}

// ---------------------------------------------------------------------------
// Fused SAGE layer GEMM (pure loads + WMMA in the K-loop):
//   acc[m,n] = bias[n] + mean[m,:] @ Wl[n,:] + x[m,:] @ Wr[n,:]
// B-fragment (32x16): lane l owns col l&15; elem e -> k = ((e&8)<<1)|((l>>4)<<3)|(e&7)
// => same k order as A (two 8-elem runs); weights stay row-major bf16.
// D (16x16 f32): lane l col = l&15; vgpr j -> row = j + (l>>4)*8.
// OUT_BF16: store bf16 (layer 1 -> h1bf) else f32 (layer 2 -> d_out).
// ---------------------------------------------------------------------------
template <int K, int NOUT, bool RELU, bool OUT_BF16>
__global__ void gs_sage_gemm_kernel(const __bf16* __restrict__ meanbf,
                                    const __bf16* __restrict__ xbf,
                                    const __bf16* __restrict__ Wlbf,
                                    const float*  __restrict__ bias,
                                    const __bf16* __restrict__ Wrbf,
                                    float*  __restrict__ outf,
                                    __bf16* __restrict__ outbf,
                                    int nrows) {
    constexpr int CTILES = NOUT / 16;
    const int wave   = threadIdx.x >> 5;   // 4 waves / block
    const int lane   = threadIdx.x & 31;
    const int laneLo = lane & 15;
    const int laneHi = lane >> 4;

    long gw = (long)blockIdx.x * 4 + wave;
    long rowTile = gw / CTILES;            // consecutive waves share rowTile
    int  colTile = (int)(gw % CTILES);

    long m0 = rowTile * 16;
    int  n0 = colTile * 16;
    if (m0 >= nrows) return;

    long m  = m0 + laneLo;
    long mc = m < nrows ? m : (long)nrows - 1;
    int  n  = n0 + laneLo;

    const __bf16* mrow = meanbf + mc * K;
    const __bf16* xrow = xbf + mc * K;
    const __bf16* wlr  = Wlbf + (long)n * K;
    const __bf16* wrr  = Wrbf + (long)n * K;

    float bn = bias[n];
    v8f acc;
#pragma unroll
    for (int j = 0; j < 8; ++j) acc[j] = bn;

#pragma unroll
    for (int k0 = 0; k0 < K; k0 += 32) {
        v16bf aM = load_a_frag(mrow, k0, laneHi);
        v16bf aX = load_a_frag(xrow, k0, laneHi);
        v16bf bL = load_a_frag(wlr, k0, laneHi);
        v16bf bR = load_a_frag(wrr, k0, laneHi);
        acc = __builtin_amdgcn_wmma_f32_16x16x32_bf16(
                  false, aM, false, bL, (short)0, acc, false, false);
        acc = __builtin_amdgcn_wmma_f32_16x16x32_bf16(
                  false, aX, false, bR, (short)0, acc, false, false);
    }

#pragma unroll
    for (int j = 0; j < 8; ++j) {
        long mm = m0 + j + laneHi * 8;
        if (mm < nrows) {
            float v = acc[j];
            if (RELU) v = fmaxf(v, 0.f);
            if (OUT_BF16) outbf[mm * NOUT + n] = f32_to_bf16_rne(v);
            else          outf [mm * NOUT + n] = v;
        }
    }
}

// ---------------------------------------------------------------------------
// Launch.  Inputs (setup_inputs order): x, edge_index, W1l, b1, W1r, W2l, b2, W2r
// Workspace: deg[N]f32 | agg[N*256]f32 (reused both layers) |
//            xbf[N*128] | h1bf[N*256] | meanbf[N*256] | 4x weight-bf16
// ---------------------------------------------------------------------------
extern "C" void kernel_launch(void* const* d_in, const int* in_sizes, int n_in,
                              void* d_out, int out_size, void* d_ws, size_t ws_size,
                              hipStream_t stream) {
    const float* x   = (const float*)d_in[0];
    const int*   ei  = (const int*)d_in[1];
    const float* W1l = (const float*)d_in[2];
    const float* b1  = (const float*)d_in[3];
    const float* W1r = (const float*)d_in[4];
    const float* W2l = (const float*)d_in[5];
    const float* b2  = (const float*)d_in[6];
    const float* W2r = (const float*)d_in[7];
    float* out = (float*)d_out;

    const long N = in_sizes[0] / GS_IN;
    const long E = in_sizes[1] / 2;
    const int* row = ei;         // edge_index[0]
    const int* col = ei + E;     // edge_index[1]

    float*  deg    = (float*)d_ws;
    float*  agg    = deg + N;                        // N*256 f32, shared
    __bf16* xbf    = (__bf16*)(agg + N * GS_HID);    // N*128
    __bf16* h1bf   = xbf + N * GS_IN;                // N*256
    __bf16* meanbf = h1bf + N * GS_HID;              // N*256 (L1 uses N*128)
    __bf16* w1lbf  = meanbf + N * GS_HID;
    __bf16* w1rbf  = w1lbf + GS_HID * GS_IN;
    __bf16* w2lbf  = w1rbf + GS_HID * GS_IN;
    __bf16* w2rbf  = w2lbf + GS_OUT * GS_HID;

    const int ZB = 4096, CB = 1024;

    // 1) zero deg + agg (contiguous)
    gs_zero_kernel<<<ZB, 256, 0, stream>>>((float4*)deg, (N * (1 + GS_HID)) / 4);

    // 2) pre-convert weights + x to bf16
    gs_cvt_bf16_kernel<<<64, 256, 0, stream>>>(W1l, w1lbf, (long)GS_HID * GS_IN / 2);
    gs_cvt_bf16_kernel<<<64, 256, 0, stream>>>(W1r, w1rbf, (long)GS_HID * GS_IN / 2);
    gs_cvt_bf16_kernel<<<64, 256, 0, stream>>>(W2l, w2lbf, (long)GS_OUT * GS_HID / 2);
    gs_cvt_bf16_kernel<<<64, 256, 0, stream>>>(W2r, w2rbf, (long)GS_OUT * GS_HID / 2);
    gs_cvt_bf16_kernel<<<CB, 256, 0, stream>>>(x, xbf, N * GS_IN / 2);

    // 3) layer-1 scatter (f32 x, builds degree), agg viewed as [N][128]
    {
        long threads = E * (GS_IN / 4);
        long blocks  = (threads + 255) / 256;
        gs_scatter_f32_kernel<GS_IN, true><<<(int)blocks, 256, 0, stream>>>(
            x, row, col, agg, deg, E);
    }

    // 4) mean1 -> bf16
    gs_mean_kernel<GS_IN><<<CB, 256, 0, stream>>>(agg, deg, meanbf, N * GS_IN / 2);

    // 5) layer-1 fused GEMM -> h1bf (ReLU, bf16 out)
    {
        long waves  = ((N + 15) / 16) * (GS_HID / 16);
        long blocks = (waves + 3) / 4;
        gs_sage_gemm_kernel<GS_IN, GS_HID, true, true><<<(int)blocks, 128, 0, stream>>>(
            meanbf, xbf, w1lbf, b1, w1rbf, nullptr, h1bf, (int)N);
    }

    // 6) re-zero agg, now viewed as [N][256]
    gs_zero_kernel<<<ZB, 256, 0, stream>>>((float4*)agg, (N * GS_HID) / 4);

    // 7) layer-2 scatter (bf16 h1 gather -> f32 atomics)
    {
        long threads = E * (GS_HID / 8);
        long blocks  = (threads + 255) / 256;
        gs_scatter_bf16_kernel<GS_HID><<<(int)blocks, 256, 0, stream>>>(
            h1bf, row, col, agg, E);
    }

    // 8) mean2 -> bf16
    gs_mean_kernel<GS_HID><<<CB, 256, 0, stream>>>(agg, deg, meanbf, N * GS_HID / 2);

    // 9) layer-2 fused GEMM -> out (f32)
    {
        long waves  = ((N + 15) / 16) * (GS_OUT / 16);
        long blocks = (waves + 3) / 4;
        gs_sage_gemm_kernel<GS_HID, GS_OUT, false, false><<<(int)blocks, 128, 0, stream>>>(
            meanbf, h1bf, w2lbf, b2, w2rbf, out, nullptr, (int)N);
    }
}
